// MultiHeadSelfAttention_42047729827855
// MI455X (gfx1250) — matrix operand order repaired and checked by reading, not compiled, not run on hardware
//
#include <hip/hip_runtime.h>

#define B_ 4
#define S_ 2048
#define D_ 1024
#define H_ 16
#define DEPTH_ 64
#define M_ (B_*S_)   // 8192 rows

typedef __attribute__((ext_vector_type(16))) _Float16 v16h;
typedef __attribute__((ext_vector_type(8)))  _Float16 v8h;
typedef __attribute__((ext_vector_type(8)))  float    v8f;

static __device__ __forceinline__ v8f wmma_f16(v16h a, v16h b, v8f c) {
  // D = A(16x32 f16) * B(32x16 f16) + C(16x16 f32)
  return __builtin_amdgcn_wmma_f32_16x16x32_f16(false, a, false, b, (short)0, c, false, false);
}

// A fragment (16x32) from row-major f16 (ld in halfs).
// Lane L: row M = L&15; halfs 0..7 <- K = 8*(L>>4)+0..7 ; halfs 8..15 <- +16.
static __device__ __forceinline__ v16h load_a_f16(const _Float16* p0, int ld) {
  int lane = threadIdx.x & 31;
  const _Float16* p = p0 + (size_t)(lane & 15) * ld + ((lane >> 4) << 3);
  v8h lo = *(const v8h*)p;
  v8h hi = *(const v8h*)(p + 16);
  v16h r;
#pragma unroll
  for (int i = 0; i < 8; ++i) { r[i] = lo[i]; r[i + 8] = hi[i]; }
  return r;
}

// A fragment (16x32) from row-major f32, converting to f16 in registers.
static __device__ __forceinline__ v16h load_a_f32(const float* p0, int ld) {
  int lane = threadIdx.x & 31;
  const float* p = p0 + (size_t)(lane & 15) * ld + ((lane >> 4) << 3);
  v16h r;
#pragma unroll
  for (int i = 0; i < 8; ++i) { r[i] = (_Float16)p[i]; r[i + 8] = (_Float16)p[16 + i]; }
  return r;
}

// B fragment (32x16) from row-major f16 [K][N] (ld in halfs).
// Lane L: col N = L&15; halfs 0..7 <- K = 8*(L>>4)+0..7 ; halfs 8..15 <- +16.
static __device__ __forceinline__ v16h load_b_f16(const _Float16* p0, int ld) {
  int lane = threadIdx.x & 31;
  const _Float16* p = p0 + (size_t)((lane >> 4) << 3) * ld + (lane & 15);
  v16h r;
#pragma unroll
  for (int i = 0; i < 8; ++i) { r[i] = p[(size_t)i * ld]; r[i + 8] = p[(size_t)(16 + i) * ld]; }
  return r;
}

// ---------------- Kernel 1: weight fp32 -> f16 conversion ----------------
__global__ void cvt_weights_kernel(const float* wq, const float* wk,
                                   const float* wv, const float* wo,
                                   _Float16* dst) {
  const float* src = (blockIdx.y == 0) ? wq
                   : (blockIdx.y == 1) ? wk
                   : (blockIdx.y == 2) ? wv : wo;
  size_t i = (size_t)blockIdx.x * blockDim.x + threadIdx.x;
  dst[(size_t)blockIdx.y * (D_ * D_) + i] = (_Float16)src[i];
}

// ---------------- Kernel 2: fused QKV projection GEMM ----------------
// grid: (D/64, M/128, 3) ; block 128 (4 waves, each wave = 32 rows x 64 cols)
// K-loop is software double-buffered: next iteration's fragments are in
// flight while the current 8 WMMAs issue.
__global__ void __launch_bounds__(128)
qkv_proj_kernel(const float* q, const float* k, const float* v,
                const float* bq, const float* bk, const float* bv,
                const _Float16* wh,          // [3][D][D] f16 (wq,wk,wv)
                _Float16* qh, _Float16* kht, _Float16* vh) {
  int wave = threadIdx.x >> 5, lane = threadIdx.x & 31;
  int which = blockIdx.z;
  const float* X    = (which == 0) ? q  : (which == 1) ? k  : v;
  const float* bias = (which == 0) ? bq : (which == 1) ? bk : bv;
  const _Float16* W = wh + (size_t)which * (D_ * D_);
  int mbase = blockIdx.y * 128 + wave * 32;
  int nbase = blockIdx.x * 64;
  const float* X0 = X + (size_t)mbase * D_;
  const float* X1 = X0 + (size_t)16 * D_;

  v8f c[8] = {};
  v16h a0 = load_a_f32(X0, D_);
  v16h a1 = load_a_f32(X1, D_);
  v16h b[4];
#pragma unroll
  for (int nf = 0; nf < 4; ++nf) b[nf] = load_b_f16(W + nbase + nf * 16, D_);

  for (int kt = 0; kt < D_ / 32; ++kt) {
    int kn = (kt + 1) * 32;
    v16h a0n, a1n, bn[4];
    if (kt < D_ / 32 - 1) {
      __builtin_prefetch(X0 + (size_t)(lane & 15) * D_ + kn + 32, 0, 3);
      a0n = load_a_f32(X0 + kn, D_);
      a1n = load_a_f32(X1 + kn, D_);
#pragma unroll
      for (int nf = 0; nf < 4; ++nf)
        bn[nf] = load_b_f16(W + (size_t)kn * D_ + nbase + nf * 16, D_);
    }
#pragma unroll
    for (int nf = 0; nf < 4; ++nf) {
      c[nf]     = wmma_f16(a0, b[nf], c[nf]);
      c[4 + nf] = wmma_f16(a1, b[nf], c[4 + nf]);
    }
    if (kt < D_ / 32 - 1) {
      a0 = a0n; a1 = a1n;
#pragma unroll
      for (int nf = 0; nf < 4; ++nf) b[nf] = bn[nf];
    }
  }

  int hi = lane >> 4, nc = lane & 15;
#pragma unroll
  for (int g = 0; g < 2; ++g) {
#pragma unroll
    for (int nf = 0; nf < 4; ++nf) {
#pragma unroll
      for (int r = 0; r < 8; ++r) {
        int row = mbase + g * 16 + r + 8 * hi;   // global M index = b*S + s
        int col = nbase + nf * 16 + nc;          // global N index = h*64 + d
        float val = c[g * 4 + nf][r] + bias[col];
        int bb = row >> 11, s = row & (S_ - 1);
        int h  = col >> 6,  d = col & 63;
        if (which == 0) {        // Q: fold in 1/sqrt(depth), layout [B,H,S,64]
          qh[(((size_t)(bb * H_ + h)) * S_ + s) * DEPTH_ + d] = (_Float16)(val * 0.125f);
        } else if (which == 1) { // K: transposed layout [B,H,64,S]
          kht[(((size_t)(bb * H_ + h)) * DEPTH_ + d) * S_ + s] = (_Float16)val;
        } else {                 // V: [B,H,S,64]
          vh[(((size_t)(bb * H_ + h)) * S_ + s) * DEPTH_ + d] = (_Float16)val;
        }
      }
    }
  }
}

// ---------------- Kernel 3: causal flash attention ----------------
// grid: (S/64, B*H) ; block 128 (4 waves, each wave = 16 query rows)
__global__ void __launch_bounds__(128)
attn_kernel(const _Float16* qh, const _Float16* kht, const _Float16* vh,
            _Float16* oh) {
  __shared__ _Float16 __attribute__((aligned(16))) ldsP[4][16 * 32];
  int wave = threadIdx.x >> 5, lane = threadIdx.x & 31;
  int hi = lane >> 4, nc = lane & 15;
  int bh = blockIdx.y;
  int qbase = blockIdx.x * 64 + wave * 16;

  const _Float16* Q  = qh  + ((size_t)bh * S_ + qbase) * DEPTH_;
  const _Float16* KT = kht + (size_t)bh * DEPTH_ * S_;
  const _Float16* V  = vh  + (size_t)bh * S_ * DEPTH_;

  v16h aq0 = load_a_f16(Q, DEPTH_);        // Q depth 0..31 (already scaled)
  v16h aq1 = load_a_f16(Q + 32, DEPTH_);   // Q depth 32..63

  v8f acc[4] = {};
  float m8[8], l8[8];
#pragma unroll
  for (int r = 0; r < 8; ++r) { m8[r] = -1e30f; l8[r] = 0.f; }

  int ntiles = (qbase + 47) >> 5;          // causal: only kv tiles <= diagonal
  for (int jt = 0; jt < ntiles; ++jt) {
    int kb = jt * 32;
    // logits tile = Q(16x64) * K^T(64x32), split into 2 col-frags x 2 K-chunks
    v8f s0 = {}, s1 = {};
    {
      v16h b00 = load_b_f16(KT + kb, S_);
      v16h b01 = load_b_f16(KT + kb + 16, S_);
      s0 = wmma_f16(aq0, b00, s0);
      s1 = wmma_f16(aq0, b01, s1);
      v16h b10 = load_b_f16(KT + (size_t)32 * S_ + kb, S_);
      v16h b11 = load_b_f16(KT + (size_t)32 * S_ + kb + 16, S_);
      s0 = wmma_f16(aq1, b10, s0);
      s1 = wmma_f16(aq1, b11, s1);
    }
    // V B-fragments issued early: latency hides under softmax VALU work
    v16h bv[4];
#pragma unroll
    for (int nf = 0; nf < 4; ++nf)
      bv[nf] = load_b_f16(V + (size_t)kb * DEPTH_ + nf * 16, DEPTH_);

    // causal mask (mask input is the strict upper triangle)
#pragma unroll
    for (int r = 0; r < 8; ++r) {
      int row = qbase + r + 8 * hi;
      if (kb + nc      > row) s0[r] = -1e30f;
      if (kb + 16 + nc > row) s1[r] = -1e30f;
    }
    // online softmax (row stats reduced across the 16-lane half)
    float p0[8], p1[8], alpha[8];
#pragma unroll
    for (int r = 0; r < 8; ++r) {
      float mx = fmaxf(s0[r], s1[r]);
#pragma unroll
      for (int ofs = 8; ofs >= 1; ofs >>= 1) mx = fmaxf(mx, __shfl_xor(mx, ofs, 32));
      float mnew = fmaxf(m8[r], mx);
      alpha[r] = __expf(m8[r] - mnew);
      p0[r] = __expf(s0[r] - mnew);
      p1[r] = __expf(s1[r] - mnew);
      float sum = p0[r] + p1[r];
#pragma unroll
      for (int ofs = 8; ofs >= 1; ofs >>= 1) sum += __shfl_xor(sum, ofs, 32);
      l8[r] = l8[r] * alpha[r] + sum;
      m8[r] = mnew;
    }
#pragma unroll
    for (int nf = 0; nf < 4; ++nf)
#pragma unroll
      for (int r = 0; r < 8; ++r) acc[nf][r] *= alpha[r];

    // P: C-layout f32 -> A-layout f16 via per-wave LDS slab (same-wave ordering)
    _Float16* Pw = &ldsP[wave][0];
#pragma unroll
    for (int r = 0; r < 8; ++r) {
      int m = r + 8 * hi;
      Pw[m * 32 + nc]      = (_Float16)p0[r];
      Pw[m * 32 + 16 + nc] = (_Float16)p1[r];
    }
    v16h ap = load_a_f16(Pw, 32);
    // O += P(16x32) * V(32x64)
#pragma unroll
    for (int nf = 0; nf < 4; ++nf)
      acc[nf] = wmma_f16(ap, bv[nf], acc[nf]);
  }

  // normalize and store O [B,H,S,64] f16
  _Float16* O = oh + ((size_t)bh * S_ + qbase) * DEPTH_;
#pragma unroll
  for (int r = 0; r < 8; ++r) {
    float inv = 1.f / l8[r];
    int row = r + 8 * hi;
#pragma unroll
    for (int nf = 0; nf < 4; ++nf)
      O[(size_t)row * DEPTH_ + nf * 16 + nc] = (_Float16)(acc[nf][r] * inv);
  }
}

// ---------------- Kernel 4: output projection GEMM (fp32 out) ----------------
// grid: (D/64, M/128) ; block 128 (wave = 32 rows x 64 cols, double-buffered)
__global__ void __launch_bounds__(128)
out_proj_kernel(const _Float16* oh, const _Float16* woh, const float* bo,
                float* out) {
  int wave = threadIdx.x >> 5, lane = threadIdx.x & 31;
  int mbase = blockIdx.y * 128 + wave * 32;
  int nbase = blockIdx.x * 64;
  int bb = mbase >> 11, sm = mbase & (S_ - 1);

  // concat-heads A source: col k -> (head = k/64, depth offset = k%64)
  const _Float16* Abase = oh + (size_t)bb * H_ * S_ * DEPTH_;

  v8f c[8] = {};
  v16h a0 = load_a_f16(Abase + ((size_t)0 * S_ + sm) * DEPTH_, DEPTH_);
  v16h a1 = load_a_f16(Abase + ((size_t)0 * S_ + sm + 16) * DEPTH_, DEPTH_);
  v16h b[4];
#pragma unroll
  for (int nf = 0; nf < 4; ++nf) b[nf] = load_b_f16(woh + nbase + nf * 16, D_);

  for (int kt = 0; kt < D_ / 32; ++kt) {
    int kn = (kt + 1) * 32;
    v16h a0n, a1n, bn[4];
    if (kt < D_ / 32 - 1) {
      int hn = kn >> 6, dofs = kn & 63;
      const _Float16* A = Abase + (((size_t)hn * S_) + sm) * DEPTH_ + dofs;
      a0n = load_a_f16(A, DEPTH_);
      a1n = load_a_f16(A + (size_t)16 * DEPTH_, DEPTH_);
#pragma unroll
      for (int nf = 0; nf < 4; ++nf)
        bn[nf] = load_b_f16(woh + (size_t)kn * D_ + nbase + nf * 16, D_);
    }
#pragma unroll
    for (int nf = 0; nf < 4; ++nf) {
      c[nf]     = wmma_f16(a0, b[nf], c[nf]);
      c[4 + nf] = wmma_f16(a1, b[nf], c[4 + nf]);
    }
    if (kt < D_ / 32 - 1) {
      a0 = a0n; a1 = a1n;
#pragma unroll
      for (int nf = 0; nf < 4; ++nf) b[nf] = bn[nf];
    }
  }

  int hi = lane >> 4, nc = lane & 15;
#pragma unroll
  for (int g = 0; g < 2; ++g)
#pragma unroll
    for (int nf = 0; nf < 4; ++nf)
#pragma unroll
      for (int r = 0; r < 8; ++r) {
        int row = mbase + g * 16 + r + 8 * hi;
        int col = nbase + nf * 16 + nc;
        out[(size_t)row * D_ + col] = c[g * 4 + nf][r] + bo[col];
      }
}

// ---------------- Host launcher ----------------
extern "C" void kernel_launch(void* const* d_in, const int* in_sizes, int n_in,
                              void* d_out, int out_size, void* d_ws, size_t ws_size,
                              hipStream_t stream) {
  const float* v  = (const float*)d_in[0];
  const float* k  = (const float*)d_in[1];
  const float* q  = (const float*)d_in[2];
  // d_in[3] = mask: strict upper triangle -> implemented analytically (causal)
  const float* wq = (const float*)d_in[4];
  const float* bq = (const float*)d_in[5];
  const float* wk = (const float*)d_in[6];
  const float* bk = (const float*)d_in[7];
  const float* wv = (const float*)d_in[8];
  const float* bv = (const float*)d_in[9];
  const float* wo = (const float*)d_in[10];
  const float* bo = (const float*)d_in[11];
  float* out = (float*)d_out;

  _Float16* ws = (_Float16*)d_ws;
  const size_t WSZ = (size_t)D_ * D_;                 // 1M halfs per weight
  const size_t TSZ = (size_t)B_ * H_ * S_ * DEPTH_;   // 8M halfs per tensor
  _Float16* wh  = ws;              // wq,wk,wv f16 (+ wo as 4th slot)
  _Float16* woh = ws + 3 * WSZ;
  _Float16* qh  = ws + 4 * WSZ;    // [B,H,S,64]  (pre-scaled by 1/8)
  _Float16* kht = qh + TSZ;        // [B,H,64,S]
  _Float16* vh  = kht + TSZ;       // [B,H,S,64]
  _Float16* oh  = vh + TSZ;        // [B,H,S,64]
  // total workspace: 36M halfs = 72 MB

  cvt_weights_kernel<<<dim3(D_ * D_ / 256, 4), 256, 0, stream>>>(wq, wk, wv, wo, wh);
  qkv_proj_kernel<<<dim3(D_ / 64, M_ / 128, 3), 128, 0, stream>>>(
      q, k, v, bq, bk, bv, wh, qh, kht, vh);
  attn_kernel<<<dim3(S_ / 64, B_ * H_), 128, 0, stream>>>(qh, kht, vh, oh);
  out_proj_kernel<<<dim3(D_ / 64, M_ / 128), 128, 0, stream>>>(oh, woh, bo, out);
}
